// MultiHeadAttention_26749056320111
// MI455X (gfx1250) — matrix operand order repaired and checked by reading, not compiled, not run on hardware
//
#include <hip/hip_runtime.h>

// MI455X / gfx1250, wave32. All matmul work through v_wmma_f32_16x16x32_bf16.
// Roofline: ~69 GFLOP total, ~150MB HBM traffic -> bf16-WMMA compute is nearly
// free; design is bandwidth/L2-locality driven (mask + per-head K/V live in L2).
// This revision hoists all fragment loads ahead of the WMMA burst so the
// WMMA->VALU WAR hazard (4 co-exec NOPs for bf16 WMMA) is paid once per
// k-step instead of once per WMMA pair, and unrolls K by 2 so the compiler can
// double-buffer fragment registers.

typedef __attribute__((ext_vector_type(16))) __bf16 v16bf;
typedef __attribute__((ext_vector_type(8)))  __bf16 v8bf;
typedef __attribute__((ext_vector_type(8)))  float  v8f;

#define kD   1024
#define kS   2048
#define kBATCH 2
#define kH   16
#define kDh  64
#define kM   (kBATCH * kS)   // 4096 rows for the projection GEMMs

__device__ __forceinline__ v8f wmma_bf16(v16bf a, v16bf b, v8f c) {
  // D = A(16x32 bf16) * B(32x16 bf16) + C(16x16 f32)
  return __builtin_amdgcn_wmma_f32_16x16x32_bf16(false, a, false, b, (short)0, c,
                                                 false, false);
}

__device__ __forceinline__ v16bf concat8(v8bf a, v8bf b) {
  return __builtin_shufflevector(a, b, 0, 1, 2, 3, 4, 5, 6, 7,
                                        8, 9, 10, 11, 12, 13, 14, 15);
}

// Convert two groups of 8 consecutive f32 into one 16-element bf16 fragment.
__device__ __forceinline__ v16bf cvt_f32x16(const float* p0, const float* p1) {
  float4 a = *(const float4*)(p0);
  float4 b = *(const float4*)(p0 + 4);
  float4 c = *(const float4*)(p1);
  float4 d = *(const float4*)(p1 + 4);
  v16bf r;
  r[0] = (__bf16)a.x;  r[1] = (__bf16)a.y;  r[2] = (__bf16)a.z;  r[3] = (__bf16)a.w;
  r[4] = (__bf16)b.x;  r[5] = (__bf16)b.y;  r[6] = (__bf16)b.z;  r[7] = (__bf16)b.w;
  r[8] = (__bf16)c.x;  r[9] = (__bf16)c.y;  r[10] = (__bf16)c.z; r[11] = (__bf16)c.w;
  r[12] = (__bf16)d.x; r[13] = (__bf16)d.y; r[14] = (__bf16)d.z; r[15] = (__bf16)d.w;
  return r;
}

__device__ __forceinline__ unsigned int pk_bf16(float lo, float hi) {
  union { __bf16 h[2]; unsigned int u; } t;
  t.h[0] = (__bf16)lo; t.h[1] = (__bf16)hi;
  return t.u;
}

// ---------------------------------------------------------------------------
// Projection GEMM: Y = X[4096x1024] * W^T[1024x1024] + bias, output bf16.
// One wave computes a 32(M)x64(N) tile: 8 WMMAs per 32-wide K step, with all
// 6 fragments loaded before the WMMA burst.
// vTranspose!=0 writes V per-head transposed: Vt[b][h][d][s].
// ---------------------------------------------------------------------------
__global__ __launch_bounds__(256) void proj_f32_bf16(
    const float* __restrict__ X, const float* __restrict__ W,
    const float* __restrict__ bias, __bf16* __restrict__ Y, int vTranspose) {
  const int lane = threadIdx.x & 31;
  const int ln = lane & 15;       // column / row-in-tile selector
  const int hw = lane >> 4;       // which half-wave
  const int cA = hw * 8;          // A fragment K-chunk base
  const int wid = blockIdx.x * 8 + (threadIdx.x >> 5);
  const int mTiles = kM / 32;     // 128
  const int mT = wid % mTiles;
  const int nT = wid / mTiles;
  const int m0 = mT * 32;
  const int n0 = nT * 64;

  v8f acc[2][4];
#pragma unroll
  for (int i = 0; i < 2; i++)
#pragma unroll
    for (int j = 0; j < 4; j++)
#pragma unroll
      for (int r = 0; r < 8; r++) acc[i][j][r] = 0.f;

#pragma unroll 2
  for (int kb = 0; kb < kD; kb += 32) {
    // ---- load ALL fragments for this k-step first (distinct registers)
    v16bf Af[2];
#pragma unroll
    for (int s = 0; s < 2; s++) {
      const float* xr = X + (size_t)(m0 + s * 16 + ln) * kD + kb;
      Af[s] = cvt_f32x16(xr + 2 * cA, xr + 2 * cA + 8);   // A: lane=row m, K chunks
    }
    v16bf Bf[4];
#pragma unroll
    for (int nt = 0; nt < 4; nt++) {
      // B fragment: lane = output column n (row of W), 16 contiguous K values
      const float* wr = W + (size_t)(n0 + nt * 16 + ln) * kD + kb + hw * 16;
      Bf[nt] = cvt_f32x16(wr, wr + 8);
    }
    // ---- WMMA burst: 8 back-to-back, no operand overwrites in between
#pragma unroll
    for (int nt = 0; nt < 4; nt++) {
      acc[0][nt] = wmma_bf16(Af[0], Bf[nt], acc[0][nt]);
      acc[1][nt] = wmma_bf16(Af[1], Bf[nt], acc[1][nt]);
    }
  }

#pragma unroll
  for (int nt = 0; nt < 4; nt++) {
    const int n = n0 + nt * 16 + ln;
    const float bv = bias[n];
#pragma unroll
    for (int s = 0; s < 2; s++) {
#pragma unroll
      for (int r = 0; r < 8; r++) {
        const int m = m0 + s * 16 + r + hw * 8;   // C layout: VGPR->M, lane->N
        const float val = acc[s][nt][r] + bv;
        if (!vTranspose) {
          Y[(size_t)m * kD + n] = (__bf16)val;
        } else {
          const int b = m >> 11, sIdx = m & (kS - 1);
          const int h = n >> 6, d = n & (kDh - 1);
          Y[((size_t)((b * kH + h) * kDh + d)) * kS + sIdx] = (__bf16)val;
        }
      }
    }
  }
}

// ---------------------------------------------------------------------------
// Output projection: f32 out = AO(bf16) * Wo^T + bo
// ---------------------------------------------------------------------------
__global__ __launch_bounds__(256) void proj_bf16_f32(
    const __bf16* __restrict__ X, const float* __restrict__ W,
    const float* __restrict__ bias, float* __restrict__ Y) {
  const int lane = threadIdx.x & 31;
  const int ln = lane & 15;
  const int hw = lane >> 4;
  const int cA = hw * 8;
  const int wid = blockIdx.x * 8 + (threadIdx.x >> 5);
  const int mTiles = kM / 32;
  const int mT = wid % mTiles;
  const int nT = wid / mTiles;
  const int m0 = mT * 32;
  const int n0 = nT * 64;

  v8f acc[2][4];
#pragma unroll
  for (int i = 0; i < 2; i++)
#pragma unroll
    for (int j = 0; j < 4; j++)
#pragma unroll
      for (int r = 0; r < 8; r++) acc[i][j][r] = 0.f;

#pragma unroll 2
  for (int kb = 0; kb < kD; kb += 32) {
    v16bf Af[2];
#pragma unroll
    for (int s = 0; s < 2; s++) {
      const __bf16* xr = X + (size_t)(m0 + s * 16 + ln) * kD + kb;
      Af[s] = concat8(*(const v8bf*)(xr + 2 * cA), *(const v8bf*)(xr + 2 * cA + 8));
    }
    v16bf Bf[4];
#pragma unroll
    for (int nt = 0; nt < 4; nt++) {
      const float* wr = W + (size_t)(n0 + nt * 16 + ln) * kD + kb + hw * 16;
      Bf[nt] = cvt_f32x16(wr, wr + 8);
    }
#pragma unroll
    for (int nt = 0; nt < 4; nt++) {
      acc[0][nt] = wmma_bf16(Af[0], Bf[nt], acc[0][nt]);
      acc[1][nt] = wmma_bf16(Af[1], Bf[nt], acc[1][nt]);
    }
  }

#pragma unroll
  for (int nt = 0; nt < 4; nt++) {
    const int n = n0 + nt * 16 + ln;
    const float bv = bias[n];
#pragma unroll
    for (int s = 0; s < 2; s++) {
#pragma unroll
      for (int r = 0; r < 8; r++) {
        const int m = m0 + s * 16 + r + hw * 8;
        Y[(size_t)m * kD + n] = acc[s][nt][r] + bv;
      }
    }
  }
}

// ---------------------------------------------------------------------------
// Flash attention. One wave owns 16 query rows of one (b,h).
// Computes S^T = K*Q^T (lane=query, VGPR=key) so softmax reductions are
// in-lane + one shfl_xor(16), and x^T = V^T * P^T so the exp'd tile feeds the
// next WMMA as a B-fragment after 8 cross-half shuffles (no LDS transpose).
// ---------------------------------------------------------------------------
__global__ __launch_bounds__(256) void flash_attn(
    const __bf16* __restrict__ Qp, const __bf16* __restrict__ Kp,
    const __bf16* __restrict__ Vt, const int* __restrict__ mask,
    __bf16* __restrict__ AO) {
  const int lane = threadIdx.x & 31;
  const int ln = lane & 15;
  const int hw = lane >> 4;
  const int cA = hw * 8;
  const int w = threadIdx.x >> 5;
  const int bh = blockIdx.x;          // 0..31
  const int b = bh >> 4, h = bh & 15;
  const int q0 = (blockIdx.y * 8 + w) * 16;

  const __bf16* Qb = Qp + (size_t)b * kS * kD + (size_t)h * kDh;
  const __bf16* Kb = Kp + (size_t)b * kS * kD + (size_t)h * kDh;
  const __bf16* Vb = Vt + (size_t)(b * kH + h) * kDh * kS;

  // Q as B-fragments (lane = query column, 16 contiguous head dims), reused
  // across the whole key loop.
  v16bf Bq0, Bq1;
  {
    const __bf16* qr = Qb + (size_t)(q0 + ln) * kD + hw * 16;
    Bq0 = *(const v16bf*)(qr);        // head dims 0..31
    Bq1 = *(const v16bf*)(qr + 32);   // head dims 32..63
  }

  float m_run = -1e30f, l_run = 0.f;
  v8f acc[4];
#pragma unroll
  for (int mt = 0; mt < 4; mt++)
#pragma unroll
    for (int r = 0; r < 8; r++) acc[mt][r] = 0.f;

  for (int k0 = 0; k0 < kS; k0 += 32) {
    // ---- load both key tiles' A-fragments, and V^T fragments, up front so
    // the score WMMAs run back-to-back and V loads hide in their shadow.
    v16bf Ka[2][2];
#pragma unroll
    for (int t = 0; t < 2; t++) {
      const __bf16* kr = Kb + (size_t)(k0 + t * 16 + ln) * kD;
      Ka[t][0] = concat8(*(const v8bf*)(kr + 2 * cA), *(const v8bf*)(kr + 2 * cA + 8));
      Ka[t][1] = concat8(*(const v8bf*)(kr + 32 + 2 * cA), *(const v8bf*)(kr + 40 + 2 * cA));
    }
    v16bf Va[4];
#pragma unroll
    for (int mt = 0; mt < 4; mt++) {
      const __bf16* vr = Vb + (size_t)(mt * 16 + ln) * kS + k0;
      Va[mt] = concat8(*(const v8bf*)(vr + 2 * cA), *(const v8bf*)(vr + 2 * cA + 8));
    }
    // ---- scores^T for 32 keys x 16 queries (two 16x16 C tiles)
    v8f st[2];
#pragma unroll
    for (int t = 0; t < 2; t++) {
      v8f s;
#pragma unroll
      for (int r = 0; r < 8; r++) s[r] = 0.f;
      s = wmma_bf16(Ka[t][0], Bq0, s);
      s = wmma_bf16(Ka[t][1], Bq1, s);
      st[t] = s;
    }
    // ---- mask + scale (mask is L2-resident; key index runs along VGPRs ->
    // contiguous int4 loads per lane)
    const int* mrow = mask + (size_t)(q0 + ln) * kS + k0 + cA;
    int4 m0v = *(const int4*)(mrow);
    int4 m1v = *(const int4*)(mrow + 4);
    int4 m2v = *(const int4*)(mrow + 16);
    int4 m3v = *(const int4*)(mrow + 20);
    int mi[16] = {m0v.x, m0v.y, m0v.z, m0v.w, m1v.x, m1v.y, m1v.z, m1v.w,
                  m2v.x, m2v.y, m2v.z, m2v.w, m3v.x, m3v.y, m3v.z, m3v.w};
    float p[16];
    float mx = -1e30f;
#pragma unroll
    for (int i = 0; i < 16; i++) {
      float sv = (i < 8) ? st[0][i] : st[1][i - 8];
      sv = mi[i] ? sv * 0.125f : -1e20f;   // 1/sqrt(64); NEG on mask==0
      p[i] = sv;
      mx = fmaxf(mx, sv);
    }
    mx = fmaxf(mx, __shfl_xor(mx, 16));    // each half holds 8 of 16 key regs
    const float m_new = fmaxf(m_run, mx);
    const float corr = __expf(m_run - m_new);
    float rs = 0.f;
#pragma unroll
    for (int i = 0; i < 16; i++) {
      p[i] = __expf(p[i] - m_new);
      rs += p[i];
    }
    rs += __shfl_xor(rs, 16);
    l_run = l_run * corr + rs;
    m_run = m_new;
#pragma unroll
    for (int mt = 0; mt < 4; mt++)
#pragma unroll
      for (int r = 0; r < 8; r++) acc[mt][r] *= corr;

    // ---- build P^T B-fragment (K=32 keys): pack key pairs, swap halves so
    // each lane holds its query's 16 consecutive keys.
    unsigned int pk0[4], pk1[4];
#pragma unroll
    for (int j = 0; j < 4; j++) {
      pk0[j] = pk_bf16(p[2 * j], p[2 * j + 1]);        // keys (hw?8:0)+2j..
      pk1[j] = pk_bf16(p[8 + 2 * j], p[9 + 2 * j]);    // keys 16+(hw?8:0)+2j..
    }
    union { unsigned int u[8]; v16bf v; } pf;
#pragma unroll
    for (int j = 0; j < 4; j++) {
      unsigned int oa = __shfl_xor(pk0[j], 16);
      unsigned int ob = __shfl_xor(pk1[j], 16);
      pf.u[j]     = hw ? ob : pk0[j];   // keys 0-7  | 16-23
      pf.u[4 + j] = hw ? pk1[j] : oa;   // keys 8-15 | 24-31
    }

    // ---- x^T += V^T * P^T  (A-fragments already resident)
#pragma unroll
    for (int mt = 0; mt < 4; mt++) {
      acc[mt] = wmma_bf16(Va[mt], pf.v, acc[mt]);
    }
  }

  const float rl = 1.0f / l_run;
  // x^T C layout: lane -> query, VGPR -> head dim. Store back to [B,S,D].
  __bf16* orow = AO + (size_t)(b * kS + q0 + ln) * kD + h * kDh;
#pragma unroll
  for (int mt = 0; mt < 4; mt++)
#pragma unroll
    for (int r = 0; r < 8; r++)
      orow[mt * 16 + r + hw * 8] = (__bf16)(acc[mt][r] * rl);
}

// ---------------------------------------------------------------------------
extern "C" void kernel_launch(void* const* d_in, const int* in_sizes, int n_in,
                              void* d_out, int out_size, void* d_ws, size_t ws_size,
                              hipStream_t stream) {
  const float* q    = (const float*)d_in[0];
  const float* k    = (const float*)d_in[1];
  const float* v    = (const float*)d_in[2];
  const int*   mask = (const int*)d_in[3];
  const float* Wq   = (const float*)d_in[4];
  const float* bq   = (const float*)d_in[5];
  const float* Wk   = (const float*)d_in[6];
  const float* bk   = (const float*)d_in[7];
  const float* Wv   = (const float*)d_in[8];
  const float* bvv  = (const float*)d_in[9];
  const float* Wo   = (const float*)d_in[10];
  const float* bo   = (const float*)d_in[11];

  const size_t elems = (size_t)kM * kD;     // 4M elems per bf16 tensor (8MB)
  __bf16* Qp = (__bf16*)d_ws;
  __bf16* Kp = Qp + elems;
  __bf16* Vt = Kp + elems;                  // per-head transposed V
  __bf16* AO = Vt + elems;                  // attention output, [B,S,D]

  dim3 blk(256);
  dim3 gemmGrid(256);                       // 2048 wave tiles / 8 waves per blk
  proj_f32_bf16<<<gemmGrid, blk, 0, stream>>>(q, Wq, bq, Qp, 0);
  proj_f32_bf16<<<gemmGrid, blk, 0, stream>>>(k, Wk, bk, Kp, 0);
  proj_f32_bf16<<<gemmGrid, blk, 0, stream>>>(v, Wv, bvv, Vt, 1);
  flash_attn<<<dim3(kBATCH * kH, kS / 128), blk, 0, stream>>>(Qp, Kp, Vt, mask, AO);
  proj_bf16_f32<<<gemmGrid, blk, 0, stream>>>(AO, Wo, bo, (float*)d_out);

  (void)in_sizes; (void)n_in; (void)out_size; (void)ws_size;
}
